// ConsecutiveSwinBlocks_86895778333416
// MI455X (gfx1250) — matrix operand order repaired and checked by reading, not compiled, not run on hardware
//
#include <hip/hip_runtime.h>

typedef __bf16 bf16;
typedef __attribute__((ext_vector_type(16))) __bf16 v16bf;
typedef __attribute__((ext_vector_type(8)))  float  v8f;
typedef __attribute__((ext_vector_type(4)))  unsigned int u32x4;
typedef __attribute__((ext_vector_type(8)))  int i32x8;
typedef __attribute__((ext_vector_type(4)))  int i32x4;

#define MPAD   352
#define NTOK   343
#define MTOT   351
#define NWIN   216
#define VOLSZ  (96*42*42*42)   // 7,112,448
#define PLANE  (42*42*42)      // 74088

#if defined(__AMDGCN__) && __has_builtin(__builtin_amdgcn_tensor_load_to_lds) && __has_builtin(__builtin_amdgcn_s_wait_tensorcnt)
#  define HAVE_TDM 1
#else
#  define HAVE_TDM 0
#endif

// ---------------- WMMA helpers ----------------

__device__ __forceinline__ v8f wmma_bf(v16bf a, v16bf b, v8f c) {
  return __builtin_amdgcn_wmma_f32_16x16x32_bf16(false, a, false, b, (short)0, c, false, false);
}

union Bf8 { uint4 u; bf16 h[8]; };
__device__ __forceinline__ Bf8 ldg8(const bf16* p){ Bf8 r; r.u = *(const uint4*)p; return r; }

// A-tile 16x32 bf16 from row-major [M][ld]: lanes 0-15 hold K 0-15, lanes 16-31 hold K 16-31
__device__ __forceinline__ v16bf load_A_bf(const bf16* base, int ld, int m0, int k0){
  int lane = threadIdx.x & 31;
  int r = lane & 15;
  int kh = (lane >> 4) << 4;              // 0 or 16
  const bf16* p = base + (size_t)(m0 + r) * ld + k0 + kh;
  Bf8 lo = ldg8(p);
  Bf8 hi = ldg8(p + 8);
  v16bf a;
#pragma unroll
  for (int i = 0; i < 8; ++i){ a[i] = lo.h[i]; a[i+8] = hi.h[i]; }
  return a;
}

// A-tile from fp32 LDS (convert to bf16 on the fly)
__device__ __forceinline__ v16bf load_A_f32(const float* base, int ld, int m0, int k0){
  int lane = threadIdx.x & 31;
  int r = lane & 15;
  int kh = (lane >> 4) << 4;
  const float* p = base + (size_t)(m0 + r) * ld + k0 + kh;
  v16bf a;
#pragma unroll
  for (int i = 0; i < 8; ++i){ a[i] = (bf16)p[i]; a[i+8] = (bf16)p[8+i]; }
  return a;
}

// B-tile from pre-swizzled tile buffer: each 32x16 tile = 32 lanes x 16 contiguous bf16 (1024 B)
__device__ __forceinline__ v16bf load_B_tile(const bf16* tiles, int tile){
  const bf16* p = tiles + ((size_t)tile * 32 + (threadIdx.x & 31)) * 16;
  Bf8 lo = ldg8(p), hi = ldg8(p + 8);
  v16bf b;
#pragma unroll
  for (int i = 0; i < 8; ++i){ b[i] = lo.h[i]; b[i+8] = hi.h[i]; }
  return b;
}

// B-tile for q@k^T: B(k,n) = K[tok0+n][col0+k]  (contiguous along k)
__device__ __forceinline__ v16bf load_B_rowT(const bf16* base, int ld, int tok0, int col0){
  int lane = threadIdx.x & 31;
  int n = lane & 15;
  int kh = (lane >> 4) << 4;
  const bf16* p = base + (size_t)(tok0 + n) * ld + col0 + kh;
  Bf8 lo = ldg8(p), hi = ldg8(p + 8);
  v16bf b;
#pragma unroll
  for (int i = 0; i < 8; ++i){ b[i] = lo.h[i]; b[i+8] = hi.h[i]; }
  return b;
}

// ---------------- TDM / LDS panel staging ----------------

__device__ __forceinline__ void stage_panel(bf16* lds, const bf16* gsrc, int cols, int rows){
#if HAVE_TDM
  if (threadIdx.x < 32){
    unsigned long long ga = (unsigned long long)(size_t)gsrc;
    u32x4 g0;
    g0[0] = 1u;                                  // count=1 (valid user descriptor)
    g0[1] = (unsigned)(size_t)(void*)lds;        // lds_addr (bytes)
    g0[2] = (unsigned)ga;                        // global_addr[31:0]
    g0[3] = (unsigned)(ga >> 32) | (2u << 30);   // global_addr[56:32] | type=2
    i32x8 g1;
    g1[0] = 1 << 16;                             // data_size=1 (2 bytes)
    g1[1] = (cols & 0xffff) << 16;               // tensor_dim0[15:0]
    g1[2] = ((unsigned)cols >> 16) | ((rows & 0xffff) << 16);   // dim0[31:16] | dim1[15:0]
    g1[3] = (((unsigned)rows >> 16) & 0xffff) | ((cols & 0xffff) << 16); // dim1[31:16] | tile_dim0
    g1[4] = rows & 0xffff;                       // tile_dim1 (tile_dim2=0)
    g1[5] = cols;                                // tensor_dim0_stride[31:0]
    g1[6] = 0;
    g1[7] = 0;
    i32x4 z = {0, 0, 0, 0};
#if __clang_major__ >= 23
    i32x8 z8 = {0, 0, 0, 0, 0, 0, 0, 0};
    __builtin_amdgcn_tensor_load_to_lds(g0, g1, z, z, z8, 0);
#else
    __builtin_amdgcn_tensor_load_to_lds(g0, g1, z, z, 0);
#endif
    __builtin_amdgcn_s_wait_tensorcnt((short)0);
  }
  __syncthreads();
#else
  const uint4* s = (const uint4*)gsrc;
  uint4* d = (uint4*)lds;
  int n16 = (cols * rows) >> 3;
  for (int i = threadIdx.x; i < n16; i += blockDim.x) d[i] = s[i];
  __syncthreads();
#endif
}

// shift-window group id for mask (dilated partition: coord = ws_idx*6 + win_idx)
__device__ __forceinline__ int grp(int t, int g1, int g2, int g3){
  int a = t / 49, b = (t / 7) % 7, c = t % 7;
  int h = a * 6 + g1, y = b * 6 + g2, z = c * 6 + g3;
  int rh = h < 35 ? 0 : (h < 39 ? 1 : 2);
  int ry = y < 35 ? 0 : (y < 39 ? 1 : 2);
  int rz = z < 35 ? 0 : (z < 39 ? 1 : 2);
  return rh * 9 + ry * 3 + rz;
}

// ---------------- kernels ----------------

// fp32 [K][N] weight -> WMMA-B tile format: tile (kt,nt) holds lane-major 16 contiguous bf16
__global__ __launch_bounds__(256) void k_pack_w(const float* __restrict__ src, bf16* __restrict__ dst,
                                                int K, int N){
  int idx = blockIdx.x * 256 + threadIdx.x;
  if (idx >= K * N) return;
  int k = idx / N, n = idx % N;
  int kt = k >> 5, kin = k & 31, khalf = kin >> 4, ke = kin & 15;
  int nt = n >> 4, nn = n & 15;
  int ntn = N >> 4;
  dst[(((size_t)(kt * ntn + nt) * 32) + nn + (khalf << 4)) * 16 + ke] = (bf16)src[idx];
}

// window partition + prompt concat + LayerNorm. one token per block (96 threads).
__global__ __launch_bounds__(96) void k_win_ln(const float* __restrict__ vol, const float* __restrict__ prompt,
                                               const float* __restrict__ g, const float* __restrict__ beta,
                                               float* __restrict__ shortb, bf16* __restrict__ hbuf, int sh){
  int wt = blockIdx.x;
  int w = wt / MPAD, t = wt % MPAD;
  int c = threadIdx.x;
  float val = 0.f;
  if (t < NTOK){
    int a = t / 49, b = (t / 7) % 7, cc = t % 7;
    int g1 = w / 36, g2 = (w / 6) % 6, g3 = w % 6;
    int h = (a * 6 + g1 + sh) % 42, y = (b * 6 + g2 + sh) % 42, z = (cc * 6 + g3 + sh) % 42;
    val = vol[(size_t)c * PLANE + h * 1764 + y * 42 + z];
  } else if (t < MTOT){
    val = prompt[(t - NTOK) * 96 + c];
  }
  __shared__ float red[96];
  __shared__ float stat[2];
  red[c] = val; __syncthreads();
  if (c < 32){
    float s = red[c] + red[c + 32] + red[c + 64];
    for (int o = 16; o > 0; o >>= 1) s += __shfl_xor(s, o);
    if (c == 0) stat[0] = s * (1.f / 96.f);
  }
  __syncthreads();
  float mu = stat[0], d = val - mu;
  red[c] = d * d; __syncthreads();
  if (c < 32){
    float s = red[c] + red[c + 32] + red[c + 64];
    for (int o = 16; o > 0; o >>= 1) s += __shfl_xor(s, o);
    if (c == 0) stat[1] = rsqrtf(s * (1.f / 96.f) + 1e-6f);
  }
  __syncthreads();
  float rs = stat[1];
  size_t idx = (size_t)wt * 96 + c;
  shortb[idx] = (t < MTOT) ? val : 0.f;
  hbuf[idx]   = (t < MTOT) ? (bf16)(d * rs * g[c] + beta[c]) : (bf16)0.f;
}

// q,k,v = h @ W + b per window. A panel staged in LDS (TDM). v stored in WMMA-B tile format.
__global__ __launch_bounds__(256) void k_qkv(const bf16* __restrict__ hbuf,
                                             const bf16* __restrict__ wq, const bf16* __restrict__ wk, const bf16* __restrict__ wv,
                                             const float* __restrict__ bq, const float* __restrict__ bk, const float* __restrict__ bv,
                                             bf16* __restrict__ qb, bf16* __restrict__ kb, bf16* __restrict__ vbT){
  __shared__ bf16 apan[MPAD * 96];
  int w = blockIdx.x;
  stage_panel(apan, hbuf + (size_t)w * (MPAD * 96), 96, MPAD);
  int wave = threadIdx.x >> 5, lane = threadIdx.x & 31;
  int n_ = lane & 15, mb = (lane >> 4) << 3;
  for (int job = wave; job < 3 * 22 * 6; job += 8){
    int mat = job / 132, rem = job % 132, mt = rem / 6, nt = rem % 6;
    const bf16* W = mat == 0 ? wq : (mat == 1 ? wk : wv);
    const float* B = mat == 0 ? bq : (mat == 1 ? bk : bv);
    v8f acc;
    float bias = B[nt * 16 + n_];
#pragma unroll
    for (int j = 0; j < 8; ++j) acc[j] = bias;
#pragma unroll
    for (int ks = 0; ks < 3; ++ks){
      v16bf a = load_A_bf(apan, 96, mt * 16, ks * 32);
      v16bf b = load_B_tile(W, ks * 6 + nt);
      acc = wmma_bf(a, b, acc);
    }
    if (mat < 2){
      bf16* Od = (mat == 0 ? qb : kb) + (size_t)w * (MPAD * 96);
#pragma unroll
      for (int j = 0; j < 8; ++j)
        Od[(size_t)(mt * 16 + mb + j) * 96 + nt * 16 + n_] = (bf16)acc[j];
    } else {
      // scatter D-tile into B-tile format: tile = (w,head, token-ktile, hd-half)
      int head = nt >> 1, hdhalf = nt & 1;
      size_t tile = ((size_t)(w * 3 + head) * 22) + (mt >> 1) * 2 + hdhalf;
      int lane2 = n_ + ((mt & 1) << 4);
      Bf8 pk;
#pragma unroll
      for (int j = 0; j < 8; ++j) pk.h[j] = (bf16)acc[j];
      *(uint4*)(vbT + tile * 512 + lane2 * 16 + mb) = pk.u;
    }
  }
}

// fused attention: one wave per (window, head, 16-query tile)
__global__ __launch_bounds__(32) void k_attn(const bf16* __restrict__ qb, const bf16* __restrict__ kb,
                                             const bf16* __restrict__ vbT, const float* __restrict__ posb,
                                             bf16* __restrict__ ob, int shifted){
  __shared__ float s[16 * MPAD];
  int id = blockIdx.x;
  int w = id / 66, rem = id % 66, head = rem / 22, qt = rem % 22;
  int lane = threadIdx.x & 31;
  int n_ = lane & 15, mb = (lane >> 4) << 3;
  int m0 = qt * 16;
  int g1 = w / 36, g2 = (w / 6) % 6, g3 = w % 6;
  const bf16* Q  = qb  + (size_t)w * (MPAD * 96) + head * 32;
  const bf16* K  = kb  + (size_t)w * (MPAD * 96) + head * 32;
  const bf16* VT = vbT + ((size_t)(w * 3 + head) * 22) * 512;
  const float scale = 0.17677669529663687f;   // 32^-0.5

  v16bf aq = load_A_bf(Q, 96, m0, 0);
  for (int nt = 0; nt < 22; ++nt){
    v16bf bk_ = load_B_rowT(K, 96, nt * 16, 0);
    v8f acc;
#pragma unroll
    for (int j = 0; j < 8; ++j) acc[j] = 0.f;
    acc = wmma_bf(aq, bk_, acc);
#pragma unroll
    for (int j = 0; j < 8; ++j){
      int row = m0 + mb + j, col = nt * 16 + n_;
      float v = acc[j] * scale;
      if (row < MTOT && col < MTOT) v += posb[(size_t)head * (MTOT * MTOT) + row * MTOT + col];
      if (col == MTOT) v = -30000.f;            // padded key column
      else if (shifted){
        if (row >= NTOK) v -= 10000.f;          // prompt query rows (discarded later)
        else if (col < NTOK && grp(row, g1, g2, g3) != grp(col, g1, g2, g3)) v -= 10000.f;
      }
      s[(mb + j) * MPAD + col] = v;
    }
  }
  __syncthreads();
  {                                             // softmax: 2 lanes per row
    int r = lane >> 1, part = lane & 1;
    float* row = s + r * MPAD + part * 176;
    float mx = -3.0e38f;
    for (int j = 0; j < 176; ++j) mx = fmaxf(mx, row[j]);
    mx = fmaxf(mx, __shfl_xor(mx, 1));
    float sum = 0.f;
    for (int j = 0; j < 176; ++j) sum += __expf(row[j] - mx);
    sum += __shfl_xor(sum, 1);
    float inv = 1.f / sum;
    for (int j = 0; j < 176; ++j) row[j] = __expf(row[j] - mx) * inv;
  }
  __syncthreads();
  v8f o0, o1;
#pragma unroll
  for (int j = 0; j < 8; ++j){ o0[j] = 0.f; o1[j] = 0.f; }
  for (int ks = 0; ks < 11; ++ks){
    v16bf a  = load_A_f32(s, MPAD, 0, ks * 32);
    v16bf b0 = load_B_tile(VT, ks * 2 + 0);
    v16bf b1 = load_B_tile(VT, ks * 2 + 1);
    o0 = wmma_bf(a, b0, o0);
    o1 = wmma_bf(a, b1, o1);
  }
  bf16* O = ob + (size_t)w * (MPAD * 96) + head * 32;
#pragma unroll
  for (int j = 0; j < 8; ++j){
    O[(size_t)(m0 + mb + j) * 96 + n_]      = (bf16)o0[j];
    O[(size_t)(m0 + mb + j) * 96 + 16 + n_] = (bf16)o1[j];
  }
}

// out-projection + residual (in place into shortb)
__global__ __launch_bounds__(256) void k_proj(const bf16* __restrict__ ob, const bf16* __restrict__ wo,
                                              const float* __restrict__ bo, float* __restrict__ shortb){
  __shared__ bf16 apan[MPAD * 96];
  int w = blockIdx.x;
  stage_panel(apan, ob + (size_t)w * (MPAD * 96), 96, MPAD);
  int wave = threadIdx.x >> 5, lane = threadIdx.x & 31;
  int n_ = lane & 15, mb = (lane >> 4) << 3;
  for (int job = wave; job < 132; job += 8){
    int mt = job / 6, nt = job % 6;
    v8f acc;
    float bias = bo[nt * 16 + n_];
#pragma unroll
    for (int j = 0; j < 8; ++j) acc[j] = bias;
#pragma unroll
    for (int ks = 0; ks < 3; ++ks){
      v16bf a = load_A_bf(apan, 96, mt * 16, ks * 32);
      v16bf b = load_B_tile(wo, ks * 6 + nt);
      acc = wmma_bf(a, b, acc);
    }
    float* S = shortb + (size_t)w * (MPAD * 96);
#pragma unroll
    for (int j = 0; j < 8; ++j){
      size_t idx = (size_t)(mt * 16 + mb + j) * 96 + nt * 16 + n_;
      S[idx] = acc[j] + S[idx];
    }
  }
}

// second LayerNorm (tokens < 343 only)
__global__ __launch_bounds__(96) void k_ln2(const float* __restrict__ shortb, const float* __restrict__ g,
                                            const float* __restrict__ beta, bf16* __restrict__ hbuf){
  int wt = blockIdx.x, t = wt % MPAD, c = threadIdx.x;
  float val = (t < NTOK) ? shortb[(size_t)wt * 96 + c] : 0.f;
  __shared__ float red[96];
  __shared__ float stat[2];
  red[c] = val; __syncthreads();
  if (c < 32){
    float s = red[c] + red[c + 32] + red[c + 64];
    for (int o = 16; o > 0; o >>= 1) s += __shfl_xor(s, o);
    if (c == 0) stat[0] = s * (1.f / 96.f);
  }
  __syncthreads();
  float mu = stat[0], d = val - mu;
  red[c] = d * d; __syncthreads();
  if (c < 32){
    float s = red[c] + red[c + 32] + red[c + 64];
    for (int o = 16; o > 0; o >>= 1) s += __shfl_xor(s, o);
    if (c == 0) stat[1] = rsqrtf(s * (1.f / 96.f) + 1e-6f);
  }
  __syncthreads();
  hbuf[(size_t)wt * 96 + c] = (t < NTOK) ? (bf16)(d * stat[1] * g[c] + beta[c]) : (bf16)0.f;
}

// MLP GEMM + residual + window-reverse scatter (undo roll)
__global__ __launch_bounds__(256) void k_mlp(const bf16* __restrict__ hbuf, const bf16* __restrict__ wm,
                                             const float* __restrict__ bm, const float* __restrict__ shortb,
                                             float* __restrict__ vol, int sh){
  __shared__ bf16 apan[MPAD * 96];
  int w = blockIdx.x;
  stage_panel(apan, hbuf + (size_t)w * (MPAD * 96), 96, MPAD);
  int wave = threadIdx.x >> 5, lane = threadIdx.x & 31;
  int n_ = lane & 15, mb = (lane >> 4) << 3;
  int g1 = w / 36, g2 = (w / 6) % 6, g3 = w % 6;
  for (int job = wave; job < 132; job += 8){
    int mt = job / 6, nt = job % 6;
    v8f acc;
    float bias = bm[nt * 16 + n_];
#pragma unroll
    for (int j = 0; j < 8; ++j) acc[j] = bias;
#pragma unroll
    for (int ks = 0; ks < 3; ++ks){
      v16bf a = load_A_bf(apan, 96, mt * 16, ks * 32);
      v16bf b = load_B_tile(wm, ks * 6 + nt);
      acc = wmma_bf(a, b, acc);
    }
#pragma unroll
    for (int j = 0; j < 8; ++j){
      int row = mt * 16 + mb + j;
      if (row < NTOK){
        int col = nt * 16 + n_;
        float v = acc[j] + shortb[(size_t)w * (MPAD * 96) + (size_t)row * 96 + col];
        int a = row / 49, b = (row / 7) % 7, cc = row % 7;
        int h = (a * 6 + g1 + sh) % 42, y = (b * 6 + g2 + sh) % 42, z = (cc * 6 + g3 + sh) % 42;
        vol[(size_t)col * PLANE + h * 1764 + y * 42 + z] = v;
      }
    }
  }
}

// patch-merge gather + LayerNorm over 768
__global__ __launch_bounds__(768) void k_merge_ln(const float* __restrict__ vol, const float* __restrict__ g,
                                                  const float* __restrict__ beta, bf16* __restrict__ mh){
  int pos = blockIdx.x, c = threadIdx.x;
  float val = 0.f;
  if (pos < 9261){
    int i = pos / 441, j = (pos / 21) % 21, k = pos % 21;
    int i2 = c / 384, j2 = (c / 192) % 2, k2 = (c / 96) % 2, ch = c % 96;
    val = vol[(size_t)ch * PLANE + (2 * i + i2) * 1764 + (2 * j + j2) * 42 + (2 * k + k2)];
  }
  __shared__ float red[768];
  __shared__ float stat[2];
  red[c] = val; __syncthreads();
  if (c < 256) red[c] += red[c + 256] + red[c + 512];
  __syncthreads();
  if (c < 64) red[c] += red[c + 64] + red[c + 128] + red[c + 192];
  __syncthreads();
  if (c < 32){
    float s = red[c] + red[c + 32];
    for (int o = 16; o > 0; o >>= 1) s += __shfl_xor(s, o);
    if (c == 0) stat[0] = s * (1.f / 768.f);
  }
  __syncthreads();
  float mu = stat[0], d = val - mu;
  red[c] = d * d; __syncthreads();
  if (c < 256) red[c] += red[c + 256] + red[c + 512];
  __syncthreads();
  if (c < 64) red[c] += red[c + 64] + red[c + 128] + red[c + 192];
  __syncthreads();
  if (c < 32){
    float s = red[c] + red[c + 32];
    for (int o = 16; o > 0; o >>= 1) s += __shfl_xor(s, o);
    if (c == 0) stat[1] = rsqrtf(s * (1.f / 768.f) + 1e-6f);
  }
  __syncthreads();
  mh[(size_t)pos * 768 + c] = (pos < 9261) ? (bf16)(d * stat[1] * g[c] + beta[c]) : (bf16)0.f;
}

// merge GEMM: (9264x768) @ (768x192) -> out[ch][pos]; A row-panel staged via TDM
__global__ __launch_bounds__(256) void k_merge_gemm(const bf16* __restrict__ mh, const bf16* __restrict__ wm,
                                                    const float* __restrict__ bm, float* __restrict__ out){
  __shared__ bf16 apan[16 * 768];
  int mt = blockIdx.x;
  stage_panel(apan, mh + (size_t)mt * 16 * 768, 768, 16);
  int wave = threadIdx.x >> 5, lane = threadIdx.x & 31;
  int n_ = lane & 15, mb = (lane >> 4) << 3;
  for (int nt = wave; nt < 12; nt += 8){
    v8f acc;
    float bias = bm[nt * 16 + n_];
#pragma unroll
    for (int j = 0; j < 8; ++j) acc[j] = bias;
    for (int ks = 0; ks < 24; ++ks){
      v16bf a = load_A_bf(apan, 768, 0, ks * 32);
      v16bf b = load_B_tile(wm, ks * 12 + nt);
      acc = wmma_bf(a, b, acc);
    }
#pragma unroll
    for (int j = 0; j < 8; ++j){
      int pos = mt * 16 + mb + j;
      if (pos < 9261) out[(size_t)(nt * 16 + n_) * 9261 + pos] = acc[j];
    }
  }
}

// ---------------- host side ----------------

extern "C" void kernel_launch(void* const* d_in, const int* in_sizes, int n_in,
                              void* d_out, int out_size, void* d_ws, size_t ws_size,
                              hipStream_t stream) {
  const float* x  = (const float*)d_in[0];
  const float* p0 = (const float*)d_in[1];
  const float* p1 = (const float*)d_in[2];
  // blk leaves: 0 ln1_g 1 ln1_b 2 wq 3 bq 4 wk 5 bk 6 wv 7 bv 8 wo 9 bo 10 ln2_g 11 ln2_b 12 mlp_w 13 mlp_b 14 pos_bias
  const float* B0[15]; const float* B1[15];
  for (int i = 0; i < 15; ++i){ B0[i] = (const float*)d_in[3 + i]; B1[i] = (const float*)d_in[18 + i]; }
  const float* mg_g = (const float*)d_in[33];
  const float* mg_b = (const float*)d_in[34];
  const float* mg_w = (const float*)d_in[35];
  const float* mg_bias = (const float*)d_in[36];

  char* ws = (char*)d_ws;
  size_t off = 0;
  auto alloc = [&](size_t bytes){ void* p = ws + off; off += (bytes + 255) & ~(size_t)255; return p; };
  float* vol1   = (float*)alloc((size_t)VOLSZ * 4);
  float* vol2   = (float*)alloc((size_t)VOLSZ * 4);
  float* shortb = (float*)alloc((size_t)NWIN * MPAD * 96 * 4);
  bf16* hbuf  = (bf16*)alloc((size_t)NWIN * MPAD * 96 * 2);
  bf16* qbuf  = (bf16*)alloc((size_t)NWIN * MPAD * 96 * 2);
  bf16* kbuf  = (bf16*)alloc((size_t)NWIN * MPAD * 96 * 2);
  bf16* vbufT = (bf16*)alloc((size_t)NWIN * 3 * 22 * 512 * 2);
  bf16* obuf  = (bf16*)alloc((size_t)NWIN * MPAD * 96 * 2);
  bf16* wbf   = (bf16*)alloc(524288);
  bf16* mh    = qbuf;   // reuse for merge activations (9264*768*2 < buffer size)

  auto run_block = [&](const float* srcvol, const float* prompt, const float* const* P,
                       float* dstvol, int sh){
    int shifted = (sh != 0);
    k_pack_w<<<36, 256, 0, stream>>>(P[2],  wbf,         96, 96);  // wq
    k_pack_w<<<36, 256, 0, stream>>>(P[4],  wbf + 9216,  96, 96);  // wk
    k_pack_w<<<36, 256, 0, stream>>>(P[6],  wbf + 18432, 96, 96);  // wv
    k_pack_w<<<36, 256, 0, stream>>>(P[8],  wbf + 27648, 96, 96);  // wo
    k_pack_w<<<36, 256, 0, stream>>>(P[12], wbf + 36864, 96, 96);  // mlp_w
    k_win_ln<<<NWIN * MPAD, 96, 0, stream>>>(srcvol, prompt, P[0], P[1], shortb, hbuf, sh);
    k_qkv<<<NWIN, 256, 0, stream>>>(hbuf, wbf, wbf + 9216, wbf + 18432,
                                    P[3], P[5], P[7], qbuf, kbuf, vbufT);
    k_attn<<<NWIN * 66, 32, 0, stream>>>(qbuf, kbuf, vbufT, P[14], obuf, shifted);
    k_proj<<<NWIN, 256, 0, stream>>>(obuf, wbf + 27648, P[9], shortb);
    k_ln2<<<NWIN * MPAD, 96, 0, stream>>>(shortb, P[10], P[11], hbuf);
    k_mlp<<<NWIN, 256, 0, stream>>>(hbuf, wbf + 36864, P[13], shortb, dstvol, sh);
  };

  run_block(x,    p0, B0, vol1, 0);
  run_block(vol1, p1, B1, vol2, 3);

  k_pack_w<<<(147456 + 255) / 256, 256, 0, stream>>>(mg_w, wbf, 768, 192);
  k_merge_ln<<<9264, 768, 0, stream>>>(vol2, mg_g, mg_b, mh);
  k_merge_gemm<<<579, 256, 0, stream>>>(mh, wbf, mg_bias, (float*)d_out);
}
